// ContrastiveAttFPConv_40381282517155
// MI455X (gfx1250) — compile-verified, matrix operand order checked by reading, not run on hardware
//
#include <hip/hip_runtime.h>
#include <hip/hip_bf16.h>
#include <math.h>

// ---------------- problem constants ----------------
#define Dnum 3
#define Unum 256
#define Hnum 8
#define DHnum 32
#define Bnum 128
#define Nnum 128
#define Enum 512
#define BN_SCALE 0.9995003746877732f   // 1/sqrt(1.001)

typedef __attribute__((ext_vector_type(16))) _Float16 v16h;
typedef __attribute__((ext_vector_type(8)))  _Float16 v8h;
typedef __attribute__((ext_vector_type(8)))  float    v8f;

// ---------------- helpers ----------------
__device__ __forceinline__ unsigned encf(float x) {
    unsigned u = __float_as_uint(x);
    return (u & 0x80000000u) ? ~u : (u | 0x80000000u);
}
__device__ __forceinline__ float decf(unsigned e) {
    unsigned u = (e & 0x80000000u) ? (e & 0x7FFFFFFFu) : ~e;
    return __uint_as_float(u);
}
__device__ __forceinline__ float sigmoidf_(float x) { return 1.0f / (1.0f + expf(-x)); }

// assemble a 16-half A fragment from two contiguous 8-half LDS chunks
__device__ __forceinline__ v16h frag_from_lds(const _Float16* p) {
    v8h lo = *(const v8h*)(p);
    v8h hi = *(const v8h*)(p + 16);
    v16h a;
#pragma unroll
    for (int i = 0; i < 8; ++i) { a[i] = lo[i]; a[i + 8] = hi[i]; }
    return a;
}

// ---------------- weight pre-pack into WMMA B-fragment layout ----------------
// Layout per ISA 7.12.2 (16-bit B 32x16): lane<16 -> N=lane, K={0..7,16..23};
// lane>=16 -> N=lane-16, K={8..15,24..31}. 16 halves stored contiguously per lane.
__global__ void pack_b_kernel(const float* __restrict__ W, _Float16* __restrict__ dst,
                              int K, int Nout) {
    int idx = blockIdx.x * blockDim.x + threadIdx.x;
    int ktiles = K >> 5, ntiles = Nout >> 4;
    if (idx >= ktiles * ntiles * 32) return;
    int lane = idx & 31;
    int tile = idx >> 5;
    int kt = tile % ktiles;
    int nt = tile / ktiles;
    int n = nt * 16 + (lane & 15);
    int kbase = kt * 32 + ((lane >> 4) ? 8 : 0);
    _Float16* out = dst + (size_t)((nt * ktiles + kt) * 32 + lane) * 16;
#pragma unroll
    for (int i = 0; i < 8; ++i) out[i]     = (_Float16)W[(size_t)(kbase + i) * Nout + n];
#pragma unroll
    for (int i = 0; i < 8; ++i) out[8 + i] = (_Float16)W[(size_t)(kbase + 16 + i) * Nout + n];
}

// ---------------- generic WMMA GEMM: C = epilogue(A[MxK] * B[KxN]) ----------------
// 256 threads = 8 waves; block computes 32 rows x 128 cols (2 m-tiles per wave,
// each B fragment feeds 2 WMMAs); grid=(M/32, Nout/128)
__global__ __launch_bounds__(256) void gemm_wmma_kernel(
    const float* __restrict__ A, const _Float16* __restrict__ Bpk,
    const float* __restrict__ bias, const float* __restrict__ gamma,
    const float* __restrict__ beta, float* __restrict__ Cout,
    int K, int Nout, int reluIn, int reluOut) {
    __shared__ _Float16 sA[32 * 256];   // K <= 256 here
    const int tid = threadIdx.x, lane = tid & 31, wave = tid >> 5;
    const int m0 = blockIdx.x * 32;
    const int ntile = blockIdx.y * 8 + wave;

    for (int i = tid; i < 32 * K; i += 256) {
        int r = i / K, c = i - r * K;
        float v = A[(size_t)(m0 + r) * K + c];
        if (reluIn) v = fmaxf(v, 0.0f);
        sA[i] = (_Float16)v;
    }
    __syncthreads();

    const int ktiles = K >> 5;
    const int r = lane & 15, sel = lane >> 4;
    v8f acc0 = {}, acc1 = {};
    const _Float16* bp = Bpk + ((size_t)ntile * ktiles) * 512 + (size_t)lane * 16;
    for (int kt = 0; kt < ktiles; ++kt) {
        const int ka = kt * 32 + (sel ? 8 : 0);
        v16h bfrag = *(const v16h*)(bp + (size_t)kt * 512);
        v16h a0 = frag_from_lds(&sA[r * K + ka]);
        v16h a1 = frag_from_lds(&sA[(16 + r) * K + ka]);
        acc0 = __builtin_amdgcn_wmma_f32_16x16x32_f16(false, a0, false, bfrag,
                                                      (short)0, acc0, false, false);
        acc1 = __builtin_amdgcn_wmma_f32_16x16x32_f16(false, a1, false, bfrag,
                                                      (short)0, acc1, false, false);
    }

    const int n = ntile * 16 + (lane & 15);
    const float bv = bias ? bias[n] : 0.0f;
    const float gs = gamma ? gamma[n] * BN_SCALE : 1.0f;
    const float bt = beta ? beta[n] : 0.0f;
#pragma unroll
    for (int rr = 0; rr < 8; ++rr) {
        float v = acc0[rr] + bv;
        if (reluOut) v = fmaxf(v, 0.0f);
        if (gamma) v = v * gs + bt;
        Cout[(size_t)(m0 + rr + (sel ? 8 : 0)) * Nout + n] = v;
        float w = acc1[rr] + bv;
        if (reluOut) w = fmaxf(w, 0.0f);
        if (gamma) w = w * gs + bt;
        Cout[(size_t)(m0 + 16 + rr + (sel ? 8 : 0)) * Nout + n] = w;
    }
}

// ---------------- fused gather-concat edge MLP: e_up = BN(ReLU(ec @ W + b)) ----------------
// A row = [nodes[src] | nodes[tgt] | edges], K=768. 32 rows/block. grid=(B*E/32, 2)
__global__ __launch_bounds__(256) void edge_mlp_kernel(
    const float* __restrict__ nodes, const float* __restrict__ edges,
    const int* __restrict__ eidx, const _Float16* __restrict__ Bpk,
    const float* __restrict__ bias, const float* __restrict__ gamma,
    const float* __restrict__ beta, float* __restrict__ eup) {
    __shared__ _Float16 sA[32 * 768];   // 48 KB
    const int tid = threadIdx.x, lane = tid & 31, wave = tid >> 5;
    const long row0 = (long)blockIdx.x * 32;

    for (int i = tid; i < 32 * 768; i += 256) {
        int r = i / 768, c = i - r * 768;
        long gr = row0 + r;
        int b = (int)(gr >> 9);           // E = 512
        float v;
        if (c < 256) {
            int s = eidx[gr * 2 + 1];
            v = nodes[((size_t)b * Nnum + s) * Unum + c];
        } else if (c < 512) {
            int t = eidx[gr * 2 + 0];
            v = nodes[((size_t)b * Nnum + t) * Unum + (c - 256)];
        } else {
            v = edges[(size_t)gr * Unum + (c - 512)];
        }
        sA[i] = (_Float16)v;
    }
    __syncthreads();

    const int ntile = blockIdx.y * 8 + wave;
    const int ktiles = 24;                 // 768/32
    const int r = lane & 15, sel = lane >> 4;
    v8f acc0 = {}, acc1 = {};
    const _Float16* bp = Bpk + ((size_t)ntile * ktiles) * 512 + (size_t)lane * 16;
#pragma unroll 4
    for (int kt = 0; kt < ktiles; ++kt) {
        const int ka = kt * 32 + (sel ? 8 : 0);
        v16h bfrag = *(const v16h*)(bp + (size_t)kt * 512);
        v16h a0 = frag_from_lds(&sA[r * 768 + ka]);
        v16h a1 = frag_from_lds(&sA[(16 + r) * 768 + ka]);
        acc0 = __builtin_amdgcn_wmma_f32_16x16x32_f16(false, a0, false, bfrag,
                                                      (short)0, acc0, false, false);
        acc1 = __builtin_amdgcn_wmma_f32_16x16x32_f16(false, a1, false, bfrag,
                                                      (short)0, acc1, false, false);
    }

    const int n = ntile * 16 + (lane & 15);
    const float bv = bias[n], gs = gamma[n] * BN_SCALE, bt = beta[n];
#pragma unroll
    for (int rr = 0; rr < 8; ++rr) {
        float v = fmaxf(acc0[rr] + bv, 0.0f) * gs + bt;
        eup[((size_t)row0 + rr + (sel ? 8 : 0)) * Unum + n] = v;
        float w = fmaxf(acc1[rr] + bv, 0.0f) * gs + bt;
        eup[((size_t)row0 + 16 + rr + (sel ? 8 : 0)) * Unum + n] = w;
    }
}

// ---------------- GAT scatter-softmax ----------------
__global__ void gat_init_kernel(unsigned* __restrict__ mmax, float* __restrict__ den,
                                float* __restrict__ att) {
    int i = blockIdx.x * 256 + threadIdx.x;
    const int BNH = Bnum * Nnum * Hnum;
    if (i < BNH) { mmax[i] = encf(-1.0e9f); den[i] = 0.0f; }
    if (i < Bnum * Nnum * Unum) att[i] = 0.0f;
}

// pass 1: logits + scatter-max. 256 threads = 32 edges x 8 heads
__global__ __launch_bounds__(256) void gat_logit_kernel(
    const float* __restrict__ h, const float* __restrict__ eup,
    const int* __restrict__ eidx, const float* __restrict__ a,
    const float* __restrict__ bgat, float* __restrict__ logit,
    unsigned* __restrict__ mmax) {
    const int t = threadIdx.x;
    const int el = t >> 3, hh = t & 7;
    const long ge = (long)blockIdx.x * 32 + el;
    const int b = (int)(ge >> 9);
    const int tgt = eidx[ge * 2 + 0], src = eidx[ge * 2 + 1];
    const float* hi = h + ((size_t)(b * Nnum + tgt)) * Unum + hh * DHnum;
    const float* hj = h + ((size_t)(b * Nnum + src)) * Unum + hh * DHnum;
    const float* ar = a + hh * (2 * DHnum + Unum);
    float s = bgat[hh];
#pragma unroll 8
    for (int dd = 0; dd < DHnum; ++dd) s += hi[dd] * ar[dd] + hj[dd] * ar[DHnum + dd];
    const float* er = eup + (size_t)ge * Unum;
    const float* a3 = ar + 2 * DHnum;
#pragma unroll 8
    for (int u = 0; u < Unum; ++u) s += er[u] * a3[u];
    s = (s > 0.0f) ? s : 0.2f * s;                    // leaky relu
    logit[ge * Hnum + hh] = s;
    atomicMax(&mmax[((size_t)(b * Nnum + tgt)) * Hnum + hh], encf(s));
}

// pass 2: exp + scatter-sum (logit buffer overwritten with ex)
__global__ void gat_exp_kernel(float* __restrict__ logit, const unsigned* __restrict__ mmax,
                               const int* __restrict__ eidx, float* __restrict__ den) {
    int idx = blockIdx.x * 256 + threadIdx.x;
    if (idx >= Bnum * Enum * Hnum) return;
    long ge = idx >> 3;
    int hh = idx & 7;
    int b = (int)(ge >> 9);
    int tgt = eidx[ge * 2 + 0];
    size_t mi = ((size_t)(b * Nnum + tgt)) * Hnum + hh;
    float ex = expf(logit[idx] - decf(mmax[mi]));
    logit[idx] = ex;
    atomicAdd(&den[mi], ex);
}

// pass 3: attended[tgt] += att * h[src]. one block per edge, 256 threads = 8 heads x 32 dims
__global__ __launch_bounds__(256) void gat_agg_kernel(
    const float* __restrict__ ex, const float* __restrict__ den,
    const float* __restrict__ h, const int* __restrict__ eidx,
    float* __restrict__ att_out) {
    const long ge = blockIdx.x;
    const int t = threadIdx.x, hh = t >> 5;
    const int b = (int)(ge >> 9);
    const int tgt = eidx[ge * 2 + 0], src = eidx[ge * 2 + 1];
    float e = ex[ge * Hnum + hh];
    float dn = den[((size_t)(b * Nnum + tgt)) * Hnum + hh] + 1e-9f;
    float v = (e / dn) * h[((size_t)(b * Nnum + src)) * Unum + t];
    atomicAdd(&att_out[((size_t)(b * Nnum + tgt)) * Unum + t], v);
}

// ---------------- GRU combine ----------------
__global__ void gru_combine_kernel(const float* __restrict__ xp, const float* __restrict__ hp,
                                   const float* __restrict__ nup, float* __restrict__ nodes) {
    int idx = blockIdx.x * 256 + threadIdx.x;
    if (idx >= Bnum * Nnum * Unum) return;
    int row = idx >> 8, u = idx & 255;
    const float* x = xp + (size_t)row * 3 * Unum;
    const float* hq = hp + (size_t)row * 3 * Unum;
    float z = sigmoidf_(x[u] + hq[u]);
    float r = sigmoidf_(x[Unum + u] + hq[Unum + u]);
    float hh = tanhf(x[2 * Unum + u] + r * hq[2 * Unum + u]);
    nodes[idx] = z * nup[idx] + (1.0f - z) * hh;
}

// ---------------- pooling / output ----------------
__global__ void mean_kernel(const float* __restrict__ nodes, float* __restrict__ emb) {
    int idx = blockIdx.x * 256 + threadIdx.x;
    if (idx >= Bnum * Unum) return;
    int b = idx >> 8, u = idx & 255;
    float s = 0.0f;
    for (int n = 0; n < Nnum; ++n) s += nodes[((size_t)b * Nnum + n) * Unum + u];
    emb[idx] = s * (1.0f / Nnum);
}

__global__ void writeout_kernel(const float* __restrict__ emb2, float* __restrict__ out) {
    int idx = blockIdx.x * 256 + threadIdx.x;
    if (idx >= Bnum * 128) return;
    float v = emb2[idx];
    int b = idx >> 7, u = idx & 127;
    out[idx] = v;                                        // emb (B,128)
    out[Bnum * 128 + ((size_t)b * 2 + 0) * 128 + u] = v; // views[:,0]
    out[Bnum * 128 + ((size_t)b * 2 + 1) * 128 + u] = v; // views[:,1]
}

// ---------------- launch ----------------
extern "C" void kernel_launch(void* const* d_in, const int* in_sizes, int n_in,
                              void* d_out, int out_size, void* d_ws, size_t ws_size,
                              hipStream_t stream) {
    const float* node_attr = (const float*)d_in[0];
    const float* edge_attr = (const float*)d_in[1];
    const int*   eidx      = (const int*)d_in[2];
    const float* W_edge = (const float*)d_in[3];
    const float* b_edge = (const float*)d_in[4];
    const float* gamma_e = (const float*)d_in[5];
    const float* beta_e  = (const float*)d_in[6];
    const float* W_node = (const float*)d_in[7];
    const float* b_node = (const float*)d_in[8];
    const float* gamma_n = (const float*)d_in[9];
    const float* beta_n  = (const float*)d_in[10];
    const float* W_gat = (const float*)d_in[11];
    const float* a_gat = (const float*)d_in[12];
    const float* b_gat = (const float*)d_in[13];
    const float* W_gru = (const float*)d_in[14];
    const float* U_gru = (const float*)d_in[15];
    const float* b_gru = (const float*)d_in[16];
    const float* W_p1 = (const float*)d_in[17];
    const float* b_p1 = (const float*)d_in[18];
    const float* W_p2 = (const float*)d_in[19];
    const float* b_p2 = (const float*)d_in[20];
    float* out = (float*)d_out;

    char* base = (char*)d_ws;
    size_t off = 0;
    auto alloc = [&](size_t bytes) -> void* {
        void* p = base + off;
        off += (bytes + 255) & ~(size_t)255;
        return p;
    };
    const size_t BNU = (size_t)Bnum * Nnum * Unum;     // 4.19M
    const size_t BEU = (size_t)Bnum * Enum * Unum;     // 16.78M
    float* f_nodes  = (float*)alloc(BNU * 4);
    float* f_edgesA = (float*)alloc(BEU * 4);
    float* f_edgesB = (float*)alloc(BEU * 4);
    float* f_h      = (float*)alloc(BNU * 4);
    float* f_nup    = (float*)alloc(BNU * 4);
    float* f_att    = (float*)alloc(BNU * 4);
    float* f_xp     = (float*)alloc(BNU * 3 * 4);
    float* f_hp     = (float*)alloc(BNU * 3 * 4);
    float* f_logit  = (float*)alloc((size_t)Bnum * Enum * Hnum * 4);
    unsigned* u_mmax = (unsigned*)alloc((size_t)Bnum * Nnum * Hnum * 4);
    float* f_den    = (float*)alloc((size_t)Bnum * Nnum * Hnum * 4);
    float* f_emb0   = (float*)alloc((size_t)Bnum * Unum * 4);
    float* f_emb1   = (float*)alloc((size_t)Bnum * Unum * 4);
    float* f_emb2   = (float*)alloc((size_t)Bnum * 128 * 4);
    _Float16* pk_Wedge = (_Float16*)alloc((size_t)Dnum * 768 * 256 * 2);
    _Float16* pk_Wnode = (_Float16*)alloc((size_t)Dnum * 256 * 256 * 2);
    _Float16* pk_Wgat  = (_Float16*)alloc((size_t)Dnum * 256 * 256 * 2);
    _Float16* pk_Wgru  = (_Float16*)alloc((size_t)Dnum * 256 * 768 * 2);
    _Float16* pk_Ugru  = (_Float16*)alloc((size_t)Dnum * 256 * 768 * 2);
    _Float16* pk_Wp1   = (_Float16*)alloc((size_t)256 * 256 * 2);
    _Float16* pk_Wp2   = (_Float16*)alloc((size_t)256 * 128 * 2);
    (void)ws_size; (void)in_sizes; (void)n_in; (void)out_size;

    // ---- pack all weights to f16 B-fragment layout ----
    auto pack = [&](const float* W, _Float16* dst, int K, int Nout) {
        int total = (K >> 5) * (Nout >> 4) * 32;
        pack_b_kernel<<<(total + 255) / 256, 256, 0, stream>>>(W, dst, K, Nout);
    };
    for (int d = 0; d < Dnum; ++d) {
        pack(W_edge + (size_t)d * 768 * 256, pk_Wedge + (size_t)d * 768 * 256, 768, 256);
        pack(W_node + (size_t)d * 256 * 256, pk_Wnode + (size_t)d * 256 * 256, 256, 256);
        pack(W_gat  + (size_t)d * 256 * 256, pk_Wgat  + (size_t)d * 256 * 256, 256, 256);
        pack(W_gru  + (size_t)d * 256 * 768, pk_Wgru  + (size_t)d * 256 * 768, 256, 768);
        pack(U_gru  + (size_t)d * 256 * 768, pk_Ugru  + (size_t)d * 256 * 768, 256, 768);
    }
    pack(W_p1, pk_Wp1, 256, 256);
    pack(W_p2, pk_Wp2, 256, 128);

    const int M_BN = Bnum * Nnum;                       // 16384
    const float* e_src[3] = { edge_attr, f_edgesA, f_edgesB };
    float* e_dst[3] = { f_edgesA, f_edgesB, f_edgesA };

    for (int d = 0; d < Dnum; ++d) {
        const float* nodes_in = (d == 0) ? node_attr : f_nodes;

        // 1) fused gather + edge MLP -> e_up
        edge_mlp_kernel<<<dim3(Bnum * Enum / 32, 2), 256, 0, stream>>>(
            nodes_in, e_src[d], eidx,
            pk_Wedge + (size_t)d * 768 * 256,
            b_edge + d * 256, gamma_e + d * 256, beta_e + d * 256, e_dst[d]);

        // 2) h = nodes @ W_gat
        gemm_wmma_kernel<<<dim3(M_BN / 32, 2), 256, 0, stream>>>(
            nodes_in, pk_Wgat + (size_t)d * 256 * 256,
            nullptr, nullptr, nullptr, f_h, 256, 256, 0, 0);

        // 3) n_up = BN(ReLU(nodes @ W_node + b))
        gemm_wmma_kernel<<<dim3(M_BN / 32, 2), 256, 0, stream>>>(
            nodes_in, pk_Wnode + (size_t)d * 256 * 256,
            b_node + d * 256, gamma_n + d * 256, beta_n + d * 256, f_nup, 256, 256, 0, 1);

        // 4) GAT scatter-softmax
        gat_init_kernel<<<(Bnum * Nnum * Unum + 255) / 256, 256, 0, stream>>>(u_mmax, f_den, f_att);
        gat_logit_kernel<<<Bnum * Enum / 32, 256, 0, stream>>>(
            f_h, e_dst[d], eidx, a_gat + (size_t)d * Hnum * (2 * DHnum + Unum),
            b_gat + d * Hnum, f_logit, u_mmax);
        gat_exp_kernel<<<(Bnum * Enum * Hnum + 255) / 256, 256, 0, stream>>>(
            f_logit, u_mmax, eidx, f_den);
        gat_agg_kernel<<<Bnum * Enum, 256, 0, stream>>>(f_logit, f_den, f_h, eidx, f_att);

        // 5) xp = ReLU(attended) @ W_gru + b_gru ; hp = n_up @ U_gru
        gemm_wmma_kernel<<<dim3(M_BN / 32, 6), 256, 0, stream>>>(
            f_att, pk_Wgru + (size_t)d * 256 * 768,
            b_gru + d * 768, nullptr, nullptr, f_xp, 256, 768, 1, 0);
        gemm_wmma_kernel<<<dim3(M_BN / 32, 6), 256, 0, stream>>>(
            f_nup, pk_Ugru + (size_t)d * 256 * 768,
            nullptr, nullptr, nullptr, f_hp, 256, 768, 0, 0);

        // 6) GRU gate combine -> new nodes
        gru_combine_kernel<<<(Bnum * Nnum * Unum + 255) / 256, 256, 0, stream>>>(
            f_xp, f_hp, f_nup, f_nodes);
    }

    // pooling head
    mean_kernel<<<(Bnum * Unum + 255) / 256, 256, 0, stream>>>(f_nodes, f_emb0);
    gemm_wmma_kernel<<<dim3(Bnum / 32, 2), 256, 0, stream>>>(
        f_emb0, pk_Wp1, b_p1, nullptr, nullptr, f_emb1, 256, 256, 0, 1);
    gemm_wmma_kernel<<<dim3(Bnum / 32, 1), 256, 0, stream>>>(
        f_emb1, pk_Wp2, b_p2, nullptr, nullptr, f_emb2, 256, 128, 0, 1);
    writeout_kernel<<<(Bnum * 128 + 255) / 256, 256, 0, stream>>>(f_emb2, out);
}
